// VectorQuantizer_36275293782016
// MI455X (gfx1250) — matrix operand order repaired and checked by reading, not compiled, not run on hardware
//
#include <hip/hip_runtime.h>
#include <hip/hip_bf16.h>

// ---------------------------------------------------------------------------
// VQ-VAE vector quantizer for MI455X (gfx1250, wave32, WMMA + async-to-LDS)
//   M = 16384 rows, C = 256 channels, K = 8192 codes
// ---------------------------------------------------------------------------

#define M_ROWS   16384
#define C_DIM    256
#define K_VOCAB  8192
#define COL_PAD  264        // shorts per LDS column (256 + 8 pad -> bank spread)

typedef __attribute__((ext_vector_type(16))) __bf16 v16bf;
typedef __attribute__((ext_vector_type(8)))  float  v8f;
typedef int v4i __attribute__((ext_vector_type(4)));
typedef v4i __attribute__((address_space(1)))* as1_v4i_p;   // global
typedef v4i __attribute__((address_space(3)))* as3_v4i_p;   // LDS

// ---- gfx1250 async global->LDS path (guarded; fallback = regular ld/st) ----
#if defined(__has_builtin)
#if __has_builtin(__builtin_amdgcn_global_load_async_to_lds_b128)
#define VQ_USE_ASYNC 1
#endif
#endif
#ifndef VQ_USE_ASYNC
#define VQ_USE_ASYNC 0
#endif

__device__ __forceinline__ void vq_async_copy16(const unsigned short* gsrc,
                                                unsigned short* ldst) {
#if VQ_USE_ASYNC
    as1_v4i_p gp = (as1_v4i_p)(unsigned long long)gsrc;           // AS1 (global)
    as3_v4i_p lp = (as3_v4i_p)(unsigned)(unsigned long long)ldst; // AS3 = low 32 bits
    __builtin_amdgcn_global_load_async_to_lds_b128(gp, lp, 0, 0);
#else
    *(uint4*)ldst = *(const uint4*)gsrc;
#endif
}

__device__ __forceinline__ void vq_wait_async0() {
#if VQ_USE_ASYNC
#if __has_builtin(__builtin_amdgcn_s_wait_asynccnt)
    __builtin_amdgcn_s_wait_asynccnt(0);
#else
    asm volatile("s_wait_asynccnt 0x0" ::: "memory");
#endif
#endif
}

__device__ __forceinline__ unsigned short f32_to_bf16_rne(float f) {
    unsigned u = __float_as_uint(f);
    unsigned r = u + 0x7FFFu + ((u >> 16) & 1u);
    return (unsigned short)(r >> 16);
}

// --------------------------- zero the histogram ----------------------------
__global__ __launch_bounds__(256) void vq_zero_counts(int* counts) {
    int i = blockIdx.x * 256 + threadIdx.x;
    if (i < K_VOCAB) counts[i] = 0;
}

// ------------- x -> bf16 copy + per-row ||x||^2 (deterministic) ------------
__global__ __launch_bounds__(256) void vq_prep_x(const float* __restrict__ x,
                                                 unsigned short* __restrict__ xb,
                                                 float* __restrict__ xsq) {
    int m = blockIdx.x;
    int c = threadIdx.x;
    float v = x[(size_t)m * C_DIM + c];
    xb[(size_t)m * C_DIM + c] = f32_to_bf16_rne(v);
    __shared__ float red[256];
    red[c] = v * v;
    __syncthreads();
    #pragma unroll
    for (int s = 128; s > 0; s >>= 1) {
        if (c < s) red[c] += red[c + s];
        __syncthreads();
    }
    if (c == 0) xsq[m] = red[0];
}

// --- emb[C][K] -> embT bf16 [K][C] + embTf32 [K][C] (coalesced reads) ------
__global__ __launch_bounds__(256) void vq_prep_emb(const float* __restrict__ emb,
                                                   unsigned short* __restrict__ embT,
                                                   float* __restrict__ embTf32) {
    int gid = blockIdx.x * 256 + threadIdx.x;   // gid = c*K + k, k fastest
    int c = gid >> 13;
    int k = gid & (K_VOCAB - 1);
    float v = emb[gid];
    embT[(size_t)k * C_DIM + c]    = f32_to_bf16_rne(v);
    embTf32[(size_t)k * C_DIM + c] = v;
}

// ---------------- per-code ||e_k||^2, fixed-order (deterministic) ----------
__global__ __launch_bounds__(256) void vq_esq(const float* __restrict__ emb,
                                              float* __restrict__ esq) {
    int k = blockIdx.x * 256 + threadIdx.x;
    float s = 0.0f;
    for (int c = 0; c < C_DIM; ++c) {
        float v = emb[(size_t)c * K_VOCAB + k];
        s = fmaf(v, v, s);
    }
    esq[k] = s;
}

// ---------------------------------------------------------------------------
// Main kernel: bf16 WMMA fused with distance argmin.
// Block = 8 waves; each wave owns a 16-row tile (A resident in VGPRs).
// All waves share each 16-column code tile, staged in LDS via async
// global->LDS copies with double buffering (ASYNCcnt pipelined).
// ---------------------------------------------------------------------------
__global__ __launch_bounds__(256) void vq_wmma_argmin(const unsigned short* __restrict__ xb,
                                                      const unsigned short* __restrict__ embT,
                                                      const float* __restrict__ xsq,
                                                      const float* __restrict__ esq,
                                                      unsigned int* __restrict__ indices,
                                                      int* __restrict__ counts) {
    // double-buffered B tile: 16 columns x 256 ch, padded to 264 shorts/col
    __shared__ unsigned short ldsB[2][16][COL_PAD];

    const int tid  = threadIdx.x;
    const int wave = tid >> 5;
    const int lane = tid & 31;
    const int h    = lane >> 4;
    const int n    = lane & 15;
    const int m0   = blockIdx.x * 128 + wave * 16;

    // ---- A fragments (16x32 bf16 per chunk), resident for whole sweep ----
    v16bf afrag[8];
    const unsigned short* xrow = xb + (size_t)(m0 + n) * C_DIM;
    #pragma unroll
    for (int t = 0; t < 8; ++t) {
        union { v16bf v; uint4 u[2]; } af;
        const int c0 = t * 32;
        af.u[0] = *(const uint4*)(xrow + c0 + h * 8);
        af.u[1] = *(const uint4*)(xrow + c0 + 16 + h * 8);
        afrag[t] = af.v;
    }

    float xs[8];
    #pragma unroll
    for (int r = 0; r < 8; ++r) xs[r] = xsq[m0 + 8 * h + r];

    float        best_d[8];
    unsigned int best_k[8];
    #pragma unroll
    for (int r = 0; r < 8; ++r) { best_d[r] = 3.4e38f; best_k[r] = 0u; }

    // This thread's two 16-byte chunks of each 8KB tile copy.
    const int id0  = tid;          // chunk ids 0..511 cover 16 cols x 512B
    const int id1  = tid + 256;
    const int col0 = id0 >> 5, off0 = (id0 & 31) * 8;
    const int col1 = id1 >> 5, off1 = (id1 & 31) * 8;

    // prologue: stage tile k0=0 into buffer 0
    vq_async_copy16(embT + (size_t)id0 * 8, &ldsB[0][col0][off0]);
    vq_async_copy16(embT + (size_t)id1 * 8, &ldsB[0][col1][off1]);
    vq_wait_async0();
    __syncthreads();

    int cur = 0;
    for (int k0 = 0; k0 < K_VOCAB; k0 += 16) {
        // kick off async copy of the next tile while we compute on `cur`
        if (k0 + 16 < K_VOCAB) {
            const unsigned short* gsrc = embT + (size_t)(k0 + 16) * C_DIM;
            vq_async_copy16(gsrc + (size_t)id0 * 8, &ldsB[cur ^ 1][col0][off0]);
            vq_async_copy16(gsrc + (size_t)id1 * 8, &ldsB[cur ^ 1][col1][off1]);
        }
        if (k0 + 32 < K_VOCAB)   // warm L2 for the tile after next
            __builtin_prefetch((const void*)(embT + (size_t)(k0 + 32) * C_DIM + id0 * 8), 0, 3);

        const unsigned int kcol = (unsigned)(k0 + n);
        const float ek = esq[kcol];
        const unsigned short* bcol = &ldsB[cur][n][h * 8];

        // Preload ALL B fragments (16x ds_load_b128 back-to-back), then run
        // the WMMA chain so LDS latency is paid once per tile, not per chunk.
        v16bf bfrag[8];
        #pragma unroll
        for (int t = 0; t < 8; ++t) {
            union { v16bf v; uint4 u[2]; } bb;
            bb.u[0] = *(const uint4*)(bcol + t * 32);       // ds_load_b128
            bb.u[1] = *(const uint4*)(bcol + t * 32 + 16);  // ds_load_b128
            bfrag[t] = bb.v;
        }

        v8f acc = {};
        #pragma unroll
        for (int t = 0; t < 8; ++t)
            acc = __builtin_amdgcn_wmma_f32_16x16x32_bf16(
                false, afrag[t], false, bfrag[t], (short)0, acc, false, false);

        #pragma unroll
        for (int r = 0; r < 8; ++r) {
            float d = xs[r] - 2.0f * acc[r] + ek;
            if (d < best_d[r] || (d == best_d[r] && kcol < best_k[r])) {
                best_d[r] = d;
                best_k[r] = kcol;
            }
        }

        vq_wait_async0();     // next tile fully landed in LDS
        __syncthreads();      // everyone done reading `cur`, copy visible
        cur ^= 1;
    }

    // Deterministic reduction across the 16 lanes of each half (same row set)
    #pragma unroll
    for (int r = 0; r < 8; ++r) {
        #pragma unroll
        for (int off = 8; off >= 1; off >>= 1) {
            float        od = __shfl_xor(best_d[r], off, 32);
            unsigned int ok = (unsigned)__shfl_xor((int)best_k[r], off, 32);
            if (od < best_d[r] || (od == best_d[r] && ok < best_k[r])) {
                best_d[r] = od;
                best_k[r] = ok;
            }
        }
    }

    if (n == 0) {   // lane 0 -> rows m0..m0+7, lane 16 -> rows m0+8..m0+15
        #pragma unroll
        for (int r = 0; r < 8; ++r) {
            int row = m0 + 8 * h + r;
            indices[row] = best_k[r];
            atomicAdd(&counts[best_k[r]], 1);   // integer atomic: deterministic
        }
    }
}

// ------ gather quantized rows (coalesced f32 embTf32) + MSE partials -------
__global__ __launch_bounds__(256) void vq_gather_loss(const float* __restrict__ x,
                                                      const float* __restrict__ embTf32,
                                                      const unsigned int* __restrict__ indices,
                                                      float* __restrict__ qout,
                                                      float* __restrict__ partials) {
    int m = blockIdx.x;
    int c = threadIdx.x;
    unsigned int k = indices[m];
    float q  = embTf32[(size_t)k * C_DIM + c];     // coalesced row
    float xv = x[(size_t)m * C_DIM + c];
    qout[(size_t)m * C_DIM + c] = q;               // straight-through == q
    float d = q - xv;
    __shared__ float red[256];
    red[c] = d * d;
    __syncthreads();
    #pragma unroll
    for (int s = 128; s > 0; s >>= 1) {
        if (c < s) red[c] += red[c + s];
        __syncthreads();
    }
    if (c == 0) partials[m] = red[0];
}

// ------------- single-block finalize: loss + perplexity (det.) -------------
__global__ __launch_bounds__(256) void vq_finalize(const float* __restrict__ partials,
                                                   const int* __restrict__ counts,
                                                   float* __restrict__ out2) {
    __shared__ float red[256];
    __shared__ float totalS;
    int t = threadIdx.x;

    float s = 0.0f;
    for (int i = t; i < M_ROWS; i += 256) s += partials[i];
    red[t] = s;
    __syncthreads();
    #pragma unroll
    for (int q = 128; q > 0; q >>= 1) {
        if (t < q) red[t] += red[t + q];
        __syncthreads();
    }
    if (t == 0) totalS = red[0];
    __syncthreads();

    float e = 0.0f;
    for (int i = t; i < K_VOCAB; i += 256) {
        float p = (float)counts[i] * (1.0f / (float)M_ROWS);
        e += p * logf(p + 1e-10f);
    }
    red[t] = e;
    __syncthreads();
    #pragma unroll
    for (int q = 128; q > 0; q >>= 1) {
        if (t < q) red[t] += red[t + q];
        __syncthreads();
    }
    if (t == 0) {
        float mse = totalS / ((float)M_ROWS * (float)C_DIM);
        out2[0] = 1.25f * mse;        // q_latent + BETA * e_latent
        out2[1] = expf(-red[0]);      // perplexity
    }
}

// ---------------------------------------------------------------------------
extern "C" void kernel_launch(void* const* d_in, const int* in_sizes, int n_in,
                              void* d_out, int out_size, void* d_ws, size_t ws_size,
                              hipStream_t stream) {
    const float* x   = (const float*)d_in[0];   // [16,1024,256] f32
    const float* emb = (const float*)d_in[1];   // [256,8192]    f32
    float* out = (float*)d_out;                 // [M*C] quantized, then loss, ppl

    char* ws = (char*)d_ws;
    unsigned short* xb       = (unsigned short*)(ws + 0);          //  8 MiB
    unsigned short* embT     = (unsigned short*)(ws + 8388608);    //  4 MiB
    float*          embTf32  = (float*)(ws + 12582912);            //  8 MiB
    float*          xsq      = (float*)(ws + 20971520);            // 64 KiB
    float*          esq      = (float*)(ws + 21037056);            // 32 KiB
    unsigned int*   indices  = (unsigned int*)(ws + 21069824);     // 64 KiB
    int*            counts   = (int*)(ws + 21135360);              // 32 KiB
    float*          partials = (float*)(ws + 21168128);            // 64 KiB

    vq_zero_counts <<<K_VOCAB / 256, 256, 0, stream>>>(counts);
    vq_prep_x      <<<M_ROWS,        256, 0, stream>>>(x, xb, xsq);
    vq_prep_emb    <<<(C_DIM * K_VOCAB) / 256, 256, 0, stream>>>(emb, embT, embTf32);
    vq_esq         <<<K_VOCAB / 256, 256, 0, stream>>>(emb, esq);
    vq_wmma_argmin <<<M_ROWS / 128,  256, 0, stream>>>(xb, embT, xsq, esq, indices, counts);
    vq_gather_loss <<<M_ROWS,        256, 0, stream>>>(x, embTf32, indices, out, partials);
    vq_finalize    <<<1,             256, 0, stream>>>(partials, counts, out + (size_t)M_ROWS * C_DIM);
}
